// AKT_88802743812521
// MI455X (gfx1250) — matrix operand order repaired and checked by reading, not compiled, not run on hardware
//
#include <hip/hip_runtime.h>
#include <cstdint>
#include <cstddef>

// ---------------- problem constants ----------------
#define BB   4
#define SS   1024
#define DDIM 256
#define HH   8
#define TOK  (BB * SS)          // 4096 tokens
#define CC   256                // concept_num

typedef __bf16 bf16;
typedef __bf16 v16bf __attribute__((ext_vector_type(16)));
typedef __bf16 v8bf  __attribute__((ext_vector_type(8)));
typedef float  v8f   __attribute__((ext_vector_type(8)));

static __device__ inline v8f wmma_bf(v16bf a, v16bf b, v8f c) {
  // D = A(16x32 bf16) * B(32x16 bf16) + C(16x16 f32)
  return __builtin_amdgcn_wmma_f32_16x16x32_bf16(false, a, false, b, (short)0, c,
                                                 false, false);
}
static __device__ inline v16bf cat8(v8bf lo, v8bf hi) {
  return __builtin_shufflevector(lo, hi, 0, 1, 2, 3, 4, 5, 6, 7,
                                 8, 9, 10, 11, 12, 13, 14, 15);
}

// ---------------------------------------------------------------------------
// Stage 1: embedding gather.  x = (concept*mu)@d_embed + concept@c_embed
//          y = r_embed[corr]*cnum + (concept*mu)@f_embed     -> bf16 copies
// one 256-thread block per token
// ---------------------------------------------------------------------------
__global__ __launch_bounds__(256) void embed_kernel(
    const int* __restrict__ inputs, const float* __restrict__ Qm,
    const float* __restrict__ c_emb, const float* __restrict__ d_emb,
    const float* __restrict__ f_emb, const float* __restrict__ mu_q,
    const float* __restrict__ r_emb, bf16* __restrict__ xb,
    bf16* __restrict__ yb) {
  const int tok = blockIdx.x;
  const int j = threadIdx.x;
  __shared__ float con[CC];
  __shared__ float conmu[CC];
  __shared__ float cnum_s;

  const int item = inputs[tok * 3 + 0] - 1;
  const int corr = inputs[tok * 3 + 2];
  const float c = Qm[(size_t)item * CC + j];
  con[j] = c;
  conmu[j] = c * mu_q[j];
  if (j == 0) cnum_s = 0.f;
  __syncthreads();
  atomicAdd(&cnum_s, c);
  __syncthreads();

  float xacc = 0.f, yacc = 0.f;
  for (int k = 0; k < CC; ++k) {
    const float cm = conmu[k], cc = con[k];
    xacc += cm * d_emb[k * DDIM + j] + cc * c_emb[k * DDIM + j];
    yacc += cm * f_emb[k * DDIM + j];
  }
  yacc += r_emb[corr * DDIM + j] * cnum_s;
  xb[(size_t)tok * DDIM + j] = (bf16)xacc;
  yb[(size_t)tok * DDIM + j] = (bf16)yacc;
}

// ---------------------------------------------------------------------------
// f32 [rows][cols] -> bf16 [cols][rows]  (so GEMM B-fragments are contiguous)
// blockIdx.z batches independent matrices.
// ---------------------------------------------------------------------------
__global__ void transpose_bf16_kernel(const float* __restrict__ in,
                                      bf16* __restrict__ out, int rows,
                                      int cols) {
  const size_t n = (size_t)rows * cols;
  in += (size_t)blockIdx.z * n;
  out += (size_t)blockIdx.z * n;
  size_t idx = (size_t)blockIdx.x * blockDim.x + threadIdx.x;
  if (idx < n) {
    int r = (int)(idx / cols), c = (int)(idx % cols);
    out[(size_t)c * rows + r] = (bf16)in[idx];
  }
}

// ---------------------------------------------------------------------------
// Generic bf16 WMMA GEMM:  C[M x N] = A[M x K] * Bt^T   (Bt stored [N][K])
// block = 128 threads (4 waves); block tile 64(M) x 64(N); wave tile 16 x 64.
// blockIdx.z batches (per-head) with strides on Bt and C.
// ---------------------------------------------------------------------------
__global__ __launch_bounds__(128) void gemm_bf16_kernel(
    const bf16* __restrict__ A, int lda, const bf16* __restrict__ Bt, int ldbt,
    long long strideB, float* __restrict__ Cf, bf16* __restrict__ Cb, int ldc,
    long long strideC, int K) {
  const int w = threadIdx.x >> 5;
  const int l = threadIdx.x & 31;
  const int m = l & 15;         // row within A-frag / col within B-frag
  const int sel = l >> 4;       // K-group select per ISA VGPR layout
  const int z = blockIdx.z;
  Bt += (size_t)z * strideB;
  const int rt = blockIdx.y * 64 + w * 16;
  const int n0 = blockIdx.x * 64;

  v8f acc[4] = {v8f{}, v8f{}, v8f{}, v8f{}};
  const bf16* arow = A + (size_t)(rt + m) * lda;
  for (int k = 0; k < K; k += 32) {
    v8bf alo = *(const v8bf*)(arow + k + sel * 16);
    v8bf ahi = *(const v8bf*)(arow + k + 8 + sel * 16);
    v16bf af = cat8(alo, ahi);
#pragma unroll
    for (int nt = 0; nt < 4; ++nt) {
      v16bf bf_ =
          *(const v16bf*)(Bt + (size_t)(n0 + nt * 16 + m) * ldbt + k + sel * 16);
      acc[nt] = wmma_bf(af, bf_, acc[nt]);
    }
  }
#pragma unroll
  for (int nt = 0; nt < 4; ++nt) {
#pragma unroll
    for (int r = 0; r < 8; ++r) {
      const int row = rt + sel * 8 + r;
      const int col = n0 + nt * 16 + m;
      const size_t ci = (size_t)z * strideC + (size_t)row * ldc + col;
      const float v = acc[nt][r];
      if (Cf) Cf[ci] = v;
      if (Cb) Cb[ci] = (bf16)v;
    }
  }
}

// ---------------------------------------------------------------------------
// Causal flash attention (masked softmax == softmax*tril/rowsum of the ref).
// grid = (S/16 query tiles, B*H); block = 128 threads (4 waves).
// Waves split the 64-key tile for Q*K^T; split 256 head dims for P*V.
// Q/K/V layout: [(h*B+b)*S + s][256] bf16.  Output AO: [b*S+s][h*256+d] bf16.
// ---------------------------------------------------------------------------
__global__ __launch_bounds__(128) void attn_kernel(
    const bf16* __restrict__ Qb, const bf16* __restrict__ Kb,
    const bf16* __restrict__ Vb, bf16* __restrict__ AO, int strict) {
  const int qt = blockIdx.x;           // query tile (16 rows)
  const int by = blockIdx.y;           // h*B + b
  const int h = by >> 2;
  const int b = by & 3;
  const size_t R0 = (size_t)by * SS;

  const int t = threadIdx.x;
  const int w = t >> 5;
  const int l = t & 31;
  const int m = l & 15;
  const int sel = l >> 4;

  __shared__ __attribute__((aligned(32))) float Sbuf[16][64];
  __shared__ __attribute__((aligned(32))) bf16 Pbuf[16][64];
  __shared__ __attribute__((aligned(32))) bf16 Vt[256][64];  // V transposed
  __shared__ float mrow[16], lrow[16], crow[16];

  if (t < 16) {
    mrow[t] = -__builtin_inff();
    lrow[t] = 0.f;
  }

  // Preload this wave's Q A-fragments for all 8 K-steps (16x256 bf16 tile)
  v16bf qa[8];
  {
    const bf16* qrow = Qb + (R0 + qt * 16 + m) * DDIM;
#pragma unroll
    for (int dc = 0; dc < 8; ++dc) {
      v8bf lo = *(const v8bf*)(qrow + dc * 32 + sel * 16);
      v8bf hi = *(const v8bf*)(qrow + dc * 32 + 8 + sel * 16);
      qa[dc] = cat8(lo, hi);
    }
  }

  v8f acc[4] = {v8f{}, v8f{}, v8f{}, v8f{}};
  const int lastkey = qt * 16 + 15 - (strict ? 1 : 0);
  const int ktn = lastkey / 64 + 1;

  for (int kt = 0; kt < ktn; ++kt) {
    const int k0 = kt * 64;
    // ---- stage V tile (64 keys x 256 dims) transposed into LDS ----
    for (int it = 0; it < 16; ++it) {
      const int a = it * 128 + t;     // 0..2047 chunks of 8 bf16
      const int key = a >> 5;
      const int ch = a & 31;
      const v8bf vv = *(const v8bf*)(Vb + (R0 + k0 + key) * DDIM + ch * 8);
#pragma unroll
      for (int e = 0; e < 8; ++e) Vt[ch * 8 + e][key] = vv[e];
    }

    // ---- scores: wave w owns keys [k0+w*16, +16) ----
    const int kw0 = k0 + w * 16;
    v8f sc = {};
#pragma unroll
    for (int dc = 0; dc < 8; ++dc) {
      v16bf kb = *(const v16bf*)(Kb + (R0 + kw0 + m) * DDIM + dc * 32 + sel * 16);
      sc = wmma_bf(qa[dc], kb, sc);
    }
    const float scale = 0.0625f;  // 1/sqrt(256)
#pragma unroll
    for (int r = 0; r < 8; ++r) {
      const int row = sel * 8 + r;
      const int qr = qt * 16 + row;
      const int kc = kw0 + m;
      const bool keep = strict ? (kc < qr) : (kc <= qr);
      Sbuf[row][w * 16 + m] = keep ? sc[r] * scale : -1e30f;
    }
    __syncthreads();

    // ---- online softmax update (16 rows, one thread each) ----
    if (t < 16) {
      const int row = t;
      float tmax = -1e30f;
      for (int c = 0; c < 64; ++c) tmax = fmaxf(tmax, Sbuf[row][c]);
      const float mo = mrow[row];
      const float mn = fmaxf(mo, tmax);
      if (mn < -1e29f) {  // fully-masked so far (only strict row 0)
        for (int c = 0; c < 64; ++c) Pbuf[row][c] = (bf16)0.f;
        crow[row] = 1.f;
      } else {
        const float cf = __expf(mo - mn);  // exp(-inf)=0 on first tile
        float sum = 0.f;
        for (int c = 0; c < 64; ++c) {
          const float p = __expf(Sbuf[row][c] - mn);
          sum += p;
          Pbuf[row][c] = (bf16)p;
        }
        mrow[row] = mn;
        lrow[row] = lrow[row] * cf + sum;
        crow[row] = cf;
      }
    }
    __syncthreads();

    // ---- rescale accumulators, then P*V (wave w owns dims [w*64, +64)) ----
    float cfv[8];
#pragma unroll
    for (int r = 0; r < 8; ++r) cfv[r] = crow[sel * 8 + r];
#pragma unroll
    for (int nt = 0; nt < 4; ++nt) {
#pragma unroll
      for (int r = 0; r < 8; ++r) acc[nt][r] *= cfv[r];
    }
#pragma unroll
    for (int kst = 0; kst < 2; ++kst) {
      v8bf plo = *(const v8bf*)&Pbuf[m][kst * 32 + sel * 16];
      v8bf phi = *(const v8bf*)&Pbuf[m][kst * 32 + 8 + sel * 16];
      v16bf pa = cat8(plo, phi);
#pragma unroll
      for (int nt = 0; nt < 4; ++nt) {
        const int dim = w * 64 + nt * 16 + m;
        v16bf vb = *(const v16bf*)&Vt[dim][kst * 32 + sel * 16];
        acc[nt] = wmma_bf(pa, vb, acc[nt]);
      }
    }
    __syncthreads();
  }

  // ---- finalize: divide by row sum (guard 0 -> matches ref guard) ----
#pragma unroll
  for (int r = 0; r < 8; ++r) {
    const int row = sel * 8 + r;
    const float lr = lrow[row];
    const float inv = (lr > 0.f) ? 1.f / lr : 0.f;
    const int qr = qt * 16 + row;
    const size_t base = ((size_t)(b * SS + qr)) * (HH * DDIM) + h * DDIM + w * 64;
#pragma unroll
    for (int nt = 0; nt < 4; ++nt)
      AO[base + nt * 16 + m] = (bf16)(acc[nt][r] * inv);
  }
}

// ---------------------------------------------------------------------------
// pred = sigmoid(concat([shift(out), x_hat]) @ d_W + d_b)
// shift == zero row s==0 of each batch.  one 256-thread block per token.
// ---------------------------------------------------------------------------
__global__ __launch_bounds__(256) void pred_kernel(
    const float* __restrict__ outf, const float* __restrict__ xhatf,
    const float* __restrict__ dW, const float* __restrict__ db,
    float* __restrict__ pred) {
  const int tok = blockIdx.x;
  const int t = threadIdx.x;
  const int s = tok & (SS - 1);
  const float o = (s == 0) ? 0.f : outf[(size_t)tok * DDIM + t];
  float v = o * dW[t] + xhatf[(size_t)tok * DDIM + t] * dW[DDIM + t];
  __shared__ float red[256];
  red[t] = v;
  __syncthreads();
  for (int st = 128; st > 0; st >>= 1) {
    if (t < st) red[t] += red[t + st];
    __syncthreads();
  }
  if (t == 0) pred[tok] = 1.f / (1.f + __expf(-(red[0] + db[0])));
}

// ---------------------------------------------------------------------------
extern "C" void kernel_launch(void* const* d_in, const int* in_sizes, int n_in,
                              void* d_out, int out_size, void* d_ws,
                              size_t ws_size, hipStream_t stream) {
  (void)in_sizes; (void)n_in; (void)out_size; (void)ws_size;
  const int* inputs  = (const int*)d_in[0];
  const float* Qm    = (const float*)d_in[1];
  const float* c_emb = (const float*)d_in[2];
  const float* d_emb = (const float*)d_in[3];
  const float* f_emb = (const float*)d_in[4];
  const float* mu_q  = (const float*)d_in[5];
  const float* r_emb = (const float*)d_in[6];
  const float* dW    = (const float*)d_in[7];
  const float* db    = (const float*)d_in[8];
  const float* wQ[3] = {(const float*)d_in[9],  (const float*)d_in[14], (const float*)d_in[19]};
  const float* wK[3] = {(const float*)d_in[10], (const float*)d_in[15], (const float*)d_in[20]};
  const float* wV[3] = {(const float*)d_in[11], (const float*)d_in[16], (const float*)d_in[21]};
  const float* wO[3] = {(const float*)d_in[12], (const float*)d_in[17], (const float*)d_in[22]};

  // ---- carve workspace (~84 MB) ----
  char* ws = (char*)d_ws;
  size_t off = 0;
  auto carve = [&](size_t bytes) -> char* {
    char* p = ws + off;
    off += (bytes + 255) & ~(size_t)255;
    return p;
  };
  bf16* xb     = (bf16*)carve((size_t)TOK * DDIM * 2);
  bf16* yb     = (bf16*)carve((size_t)TOK * DDIM * 2);
  bf16* xhat_b = (bf16*)carve((size_t)TOK * DDIM * 2);
  bf16* yhat_b = (bf16*)carve((size_t)TOK * DDIM * 2);
  float* xhat_f = (float*)carve((size_t)TOK * DDIM * 4);
  float* out_f  = (float*)carve((size_t)TOK * DDIM * 4);
  bf16* wqT = (bf16*)carve((size_t)HH * DDIM * DDIM * 2);
  bf16* wkT = (bf16*)carve((size_t)HH * DDIM * DDIM * 2);
  bf16* wvT = (bf16*)carve((size_t)HH * DDIM * DDIM * 2);
  bf16* woT = (bf16*)carve((size_t)HH * DDIM * DDIM * 2);  // 2048x256 -> [256][2048]
  bf16* Qbuf = (bf16*)carve((size_t)HH * TOK * DDIM * 2);
  bf16* Kbuf = (bf16*)carve((size_t)HH * TOK * DDIM * 2);
  bf16* Vbuf = (bf16*)carve((size_t)HH * TOK * DDIM * 2);
  bf16* AO   = (bf16*)carve((size_t)TOK * HH * DDIM * 2);

  // ---- stage 1: embeddings ----
  embed_kernel<<<TOK, 256, 0, stream>>>(inputs, Qm, c_emb, d_emb, f_emb, mu_q,
                                        r_emb, xb, yb);

  const long long hstrideW = (long long)DDIM * DDIM;   // per-head weight stride
  const long long hstrideC = (long long)TOK * DDIM;    // per-head QKV stride

  auto run_layer = [&](const bf16* in_qk, const bf16* in_v, int li,
                       float* outF, bf16* outB, int strict) {
    // weights f32 -> bf16 transposed [N][K]
    dim3 tb(256);
    transpose_bf16_kernel<<<dim3(256, 1, HH), tb, 0, stream>>>(wQ[li], wqT, DDIM, DDIM);
    transpose_bf16_kernel<<<dim3(256, 1, HH), tb, 0, stream>>>(wK[li], wkT, DDIM, DDIM);
    transpose_bf16_kernel<<<dim3(256, 1, HH), tb, 0, stream>>>(wV[li], wvT, DDIM, DDIM);
    transpose_bf16_kernel<<<dim3((HH * DDIM * DDIM + 255) / 256, 1, 1), tb, 0,
                            stream>>>(wO[li], woT, HH * DDIM, DDIM);
    // Q/K/V projections: M=4096, N=256, K=256, batched over 8 heads
    dim3 g(DDIM / 64, TOK / 64, HH), blk(128);
    gemm_bf16_kernel<<<g, blk, 0, stream>>>(in_qk, DDIM, wqT, DDIM, hstrideW,
                                            nullptr, Qbuf, DDIM, hstrideC, DDIM);
    gemm_bf16_kernel<<<g, blk, 0, stream>>>(in_qk, DDIM, wkT, DDIM, hstrideW,
                                            nullptr, Kbuf, DDIM, hstrideC, DDIM);
    gemm_bf16_kernel<<<g, blk, 0, stream>>>(in_v, DDIM, wvT, DDIM, hstrideW,
                                            nullptr, Vbuf, DDIM, hstrideC, DDIM);
    // causal attention -> AO [tok][h*256+d]
    attn_kernel<<<dim3(SS / 16, BB * HH), dim3(128), 0, stream>>>(Qbuf, Kbuf,
                                                                  Vbuf, AO,
                                                                  strict);
    // output projection: M=4096, N=256, K=2048
    gemm_bf16_kernel<<<dim3(DDIM / 64, TOK / 64, 1), blk, 0, stream>>>(
        AO, HH * DDIM, woT, HH * DDIM, 0LL, outF, outB, DDIM, 0LL, HH * DDIM);
  };

  run_layer(xb, xb, 0, xhat_f, xhat_b, 0);          // question encoder
  run_layer(yb, yb, 1, nullptr, yhat_b, 0);         // knowledge encoder
  run_layer(xhat_b, yhat_b, 2, out_f, nullptr, 1);  // knowledge retriever

  pred_kernel<<<TOK, 256, 0, stream>>>(out_f, xhat_f, dW, db, (float*)d_out);
}
